// CharEmbedding_77094662963795
// MI455X (gfx1250) — matrix operand order repaired and checked by reading, not compiled, not run on hardware
//
#include <hip/hip_runtime.h>
#include <math.h>

// Problem constants (from reference)
#define VOCAB   128
#define E_DIM   768
#define H_DIM   768
#define OUT_DIM 768
#define B_DIM   128
#define T_DIM   512

// Kernel configuration
#define NBLK     24          // persistent blocks; 24*32 = 768 channels
#define CH       32          // hidden channels per block
#define NTHREADS 256         // 8 wave32 waves

// W_hh fragments for one block, staged in LDS: 144 frag-blocks x 512 bf16 = 147456 B
#define WPK_ELEMS (144 * 512)
#define WPK_BYTES (WPK_ELEMS * 2)

typedef __attribute__((ext_vector_type(16))) __bf16 v16bf;
typedef __attribute__((ext_vector_type(8)))  __bf16 v8bf;
typedef __attribute__((ext_vector_type(8)))  float  v8f;

// ---- workspace layout (bytes) ----
#define OFF_GX   0                                        // [NBLK][VOCAB][96] f32
#define OFF_HBF  (OFF_GX  + (size_t)NBLK*VOCAB*96*4)      // [2][B][H] bf16 ping-pong
#define OFF_HF   (OFF_HBF + (size_t)2*B_DIM*H_DIM*2)      // [2][B][H] f32 ping-pong
#define OFF_BAR  (OFF_HF  + (size_t)2*B_DIM*H_DIM*4)      // barrier counter

__device__ __forceinline__ unsigned short f32_to_bf16(float f) {
  unsigned u = __builtin_bit_cast(unsigned, f);
  u += 0x7FFFu + ((u >> 16) & 1u);          // round-to-nearest-even
  return (unsigned short)(u >> 16);
}

// fast sigmoid / tanh using hardware exp + rcp (TRANS ops, co-execute with WMMA)
__device__ __forceinline__ float fast_sigmoid(float xv) {
  return __builtin_amdgcn_rcpf(1.0f + __expf(-xv));
}
__device__ __forceinline__ float fast_tanh(float xv) {
  // 1 - 2/(e^{2x}+1); exp->inf => rcp->0 => +1, exp->0 => -1 (saturates correctly)
  return 1.0f - 2.0f * __builtin_amdgcn_rcpf(__expf(2.0f * xv) + 1.0f);
}

__device__ __forceinline__ void grid_sync(unsigned* bar, unsigned target) {
  __threadfence();                           // agent-scope release (global_wb)
  __syncthreads();
  if (threadIdx.x == 0) {
    __hip_atomic_fetch_add(bar, 1u, __ATOMIC_RELEASE, __HIP_MEMORY_SCOPE_AGENT);
    while (__hip_atomic_load(bar, __ATOMIC_ACQUIRE, __HIP_MEMORY_SCOPE_AGENT) < target) {
      __builtin_amdgcn_s_sleep(1);
    }
  }
  __syncthreads();
  __threadfence();                           // agent-scope acquire (global_inv)
}

__global__ void init_barrier(unsigned* bar) { *bar = 0u; }

__global__ void __launch_bounds__(NTHREADS, 1)
gru_persistent(const int*   __restrict__ x,      // [B,T] int32
               const float* __restrict__ emb,    // [VOCAB,E]
               const float* __restrict__ W_ih,   // [3H,E]
               const float* __restrict__ W_hh,   // [3H,H]
               const float* __restrict__ b_ih,   // [3H]
               const float* __restrict__ b_hh,   // [3H]
               const float* __restrict__ fc_W,   // [OUT,H]
               const float* __restrict__ fc_b,   // [OUT]
               float*       __restrict__ out,    // [B,OUT]
               unsigned char* __restrict__ ws)
{
  extern __shared__ unsigned char dynsmem[];
  unsigned short* wpk = (unsigned short*)dynsmem;   // [144][512] bf16 B-fragments

  const int blk = blockIdx.x;
  const int tid = threadIdx.x;

  float*          gx_all = (float*)(ws + OFF_GX);
  unsigned short* hbf    = (unsigned short*)(ws + OFF_HBF);
  float*          hf     = (float*)(ws + OFF_HF);
  unsigned*       bar    = (unsigned*)(ws + OFF_BAR);

  float* gx_blk = gx_all + (size_t)blk * (VOCAB * 96);

  // ================= Phase 0: precompute =================
  // (a) gx table: gx[v][c] = emb[v,:] . W_ih[row_c,:] + b_ih[row_c]
  for (int i = tid; i < VOCAB * 96; i += NTHREADS) {
    int v = i / 96, c = i % 96;
    int g = c >> 5, jl = c & 31;
    int row = g * H_DIM + blk * CH + jl;
    const float* er = emb  + (size_t)v   * E_DIM;
    const float* wr = W_ih + (size_t)row * E_DIM;
    float acc = b_ih[row];
    for (int e = 0; e < E_DIM; ++e) acc = fmaf(er[e], wr[e], acc);
    gx_blk[i] = acc;
  }
  // (b) pack W_hh rows for this block into WMMA B-fragment order, directly in LDS:
  //     frag (g, jh, kt): lane holds W row (g*H + blk*CH + jh*16 + lane%16),
  //     k = kt*32 + (lane&16 ? 8 : 0) + (e&7) + (e&8 ? 16 : 0)  -> one contiguous 32B chunk per lane.
  for (int i = tid; i < WPK_ELEMS; i += NTHREADS) {
    int fidx = i >> 9, slot = i & 511;
    int lane = slot >> 4, e = slot & 15;
    int g = fidx / 48, rem = fidx % 48;
    int jh = rem / 24, kt = rem % 24;
    int row = g * H_DIM + blk * CH + jh * 16 + (lane & 15);
    int col = kt * 32 + ((lane & 16) ? 8 : 0) + (e & 7) + ((e & 8) ? 16 : 0);
    wpk[i] = f32_to_bf16(W_hh[(size_t)row * H_DIM + col]);
  }
  // (c) h0 = 0 (this block's channel slice, buffer 0 only)
  for (int i = tid; i < B_DIM * CH; i += NTHREADS) {
    int b = i >> 5, jl = i & 31;
    int j = blk * CH + jl;
    hf[(size_t)b * H_DIM + j]  = 0.0f;
    hbf[(size_t)b * H_DIM + j] = 0;
  }

  unsigned target = NBLK;
  grid_sync(bar, target);     // also makes LDS packing visible block-wide

  // ================= Recurrence: 512 steps, WMMA bf16, fp32 accum =================
  const int lane = tid & 31;
  const int wave = tid >> 5;
  const int mg   = wave >> 1;                 // m-tile group: batches [mg*32, mg*32+32)
  const int jh   = wave & 1;                  // channel half: 16 channels
  const int l15  = lane & 15;
  const int lhi8 = (lane & 16) ? 8 : 0;
  const int jg   = blk * CH + jh * 16 + l15;  // this lane's global hidden channel

  const float bhh_r = b_hh[0 * H_DIM + jg];
  const float bhh_z = b_hh[1 * H_DIM + jg];
  const float bhh_n = b_hh[2 * H_DIM + jg];

  const int arow0 = ((mg * 2 + 0) * 16 + l15) * H_DIM;  // A-frag row, m-tile 0
  const int arow1 = ((mg * 2 + 1) * 16 + l15) * H_DIM;  // A-frag row, m-tile 1
  const int c0    = jh * 16 + l15;                      // gx column within block slice
  // LDS base offsets for this wave's three gate fragments (elements)
  const int wb0 = ((0 * 2 + jh) * 24) * 512 + lane * 16;
  const int wb1 = ((1 * 2 + jh) * 24) * 512 + lane * 16;
  const int wb2 = ((2 * 2 + jh) * 24) * 512 + lane * 16;

  for (int t = 0; t < T_DIM; ++t) {
    const int p = t & 1;
    const unsigned short* hb_in  = hbf + (size_t)p       * (B_DIM * H_DIM);
    unsigned short*       hb_out = hbf + (size_t)(1 - p) * (B_DIM * H_DIM);
    const float*          hf_in  = hf  + (size_t)p       * (B_DIM * H_DIM);
    float*                hf_out = hf  + (size_t)(1 - p) * (B_DIM * H_DIM);

    // ---- issue all gate-input loads first: they are independent of the GEMM,
    //      so their latency hides under the whole 24-iteration WMMA chain ----
    float gxr[16], gxz[16], gxn[16], hold[16];
#pragma unroll
    for (int e2 = 0; e2 < 16; ++e2) {
      int mt = e2 >> 3, v = e2 & 7;
      int b  = (mg * 2 + mt) * 16 + v + lhi8;
      int xi = x[b * T_DIM + t];
      const float* gxp = gx_blk + (size_t)xi * 96;
      gxr[e2]  = gxp[c0];
      gxz[e2]  = gxp[32 + c0];
      gxn[e2]  = gxp[64 + c0];
      hold[e2] = hf_in[(size_t)b * H_DIM + jg];
    }

    // accumulators: 2 m-tiles x {r,z,n}; seed with b_hh
    v8f acc[2][3];
#pragma unroll
    for (int mt = 0; mt < 2; ++mt) {
#pragma unroll
      for (int q = 0; q < 8; ++q) {
        acc[mt][0][q] = bhh_r; acc[mt][1][q] = bhh_z; acc[mt][2][q] = bhh_n;
      }
    }

    // ---- GEMM k-loop: A from global (double-buffered), B from LDS ----
    v16bf aN0, aN1;
    {
      const unsigned short* pa0 = hb_in + arow0 + lhi8;
      const unsigned short* pa1 = hb_in + arow1 + lhi8;
      v8bf lo0 = *(const v8bf*)pa0, hi0 = *(const v8bf*)(pa0 + 16);
      v8bf lo1 = *(const v8bf*)pa1, hi1 = *(const v8bf*)(pa1 + 16);
      aN0 = __builtin_shufflevector(lo0, hi0, 0,1,2,3,4,5,6,7,8,9,10,11,12,13,14,15);
      aN1 = __builtin_shufflevector(lo1, hi1, 0,1,2,3,4,5,6,7,8,9,10,11,12,13,14,15);
    }
#pragma unroll
    for (int kt = 0; kt < 24; ++kt) {
      v16bf aC0 = aN0, aC1 = aN1;
      if (kt < 23) {
        const unsigned short* pa0 = hb_in + arow0 + (kt + 1) * 32 + lhi8;
        const unsigned short* pa1 = hb_in + arow1 + (kt + 1) * 32 + lhi8;
        v8bf lo0 = *(const v8bf*)pa0, hi0 = *(const v8bf*)(pa0 + 16);
        v8bf lo1 = *(const v8bf*)pa1, hi1 = *(const v8bf*)(pa1 + 16);
        aN0 = __builtin_shufflevector(lo0, hi0, 0,1,2,3,4,5,6,7,8,9,10,11,12,13,14,15);
        aN1 = __builtin_shufflevector(lo1, hi1, 0,1,2,3,4,5,6,7,8,9,10,11,12,13,14,15);
      }
#pragma unroll
      for (int g = 0; g < 3; ++g) {
        const int wb = (g == 0 ? wb0 : (g == 1 ? wb1 : wb2)) + kt * 512;
        v8bf lo = *(const v8bf*)(wpk + wb);
        v8bf hi = *(const v8bf*)(wpk + wb + 8);
        v16bf bfrag = __builtin_shufflevector(lo, hi,
            0,1,2,3,4,5,6,7,8,9,10,11,12,13,14,15);
        acc[0][g] = __builtin_amdgcn_wmma_f32_16x16x32_bf16(
            false, aC0, false, bfrag, (short)0, acc[0][g], false, false);
        acc[1][g] = __builtin_amdgcn_wmma_f32_16x16x32_bf16(
            false, aC1, false, bfrag, (short)0, acc[1][g], false, false);
      }
    }

    // ---- gate fusion + state update (all inputs already in registers) ----
#pragma unroll
    for (int mt = 0; mt < 2; ++mt) {
#pragma unroll
      for (int v = 0; v < 8; ++v) {
        int e2 = mt * 8 + v;
        int b  = (mg * 2 + mt) * 16 + v + lhi8;   // D layout: M = v (+8 upper lanes)
        float hr = acc[mt][0][v], hz = acc[mt][1][v], hn = acc[mt][2][v];
        float r = fast_sigmoid(gxr[e2] + hr);
        float z = fast_sigmoid(gxz[e2] + hz);
        float n = fast_tanh(gxn[e2] + r * hn);
        float hnew = (1.0f - z) * n + z * hold[e2];
        hf_out[(size_t)b * H_DIM + jg] = hnew;
        hb_out[(size_t)b * H_DIM + jg] = f32_to_bf16(hnew);
      }
    }

    target += NBLK;
    grid_sync(bar, target);
  }

  // ================= FC: out = h_last @ fc_W^T + fc_b (fp32, tiny) ================
  // T even -> final state lives in buffer 0.
  const float* hl = hf;
  for (int i = tid; i < B_DIM * CH; i += NTHREADS) {
    int b = i >> 5, ol = i & 31;
    int o = blk * CH + ol;
    const float* hrow = hl   + (size_t)b * H_DIM;
    const float* wrow = fc_W + (size_t)o * H_DIM;
    float acc = fc_b[o];
    for (int e = 0; e < H_DIM; ++e) acc = fmaf(hrow[e], wrow[e], acc);
    out[(size_t)b * OUT_DIM + o] = acc;
  }
}

extern "C" void kernel_launch(void* const* d_in, const int* in_sizes, int n_in,
                              void* d_out, int out_size, void* d_ws, size_t ws_size,
                              hipStream_t stream) {
  (void)in_sizes; (void)n_in; (void)out_size; (void)ws_size;
  const int*   x    = (const int*)  d_in[0];
  const float* emb  = (const float*)d_in[1];
  const float* W_ih = (const float*)d_in[2];
  const float* W_hh = (const float*)d_in[3];
  const float* b_ih = (const float*)d_in[4];
  const float* b_hh = (const float*)d_in[5];
  const float* fc_W = (const float*)d_in[6];
  const float* fc_b = (const float*)d_in[7];
  unsigned char* ws = (unsigned char*)d_ws;

  // allow >64KB dynamic LDS (CDNA5 WGP has 320KB); harmless if already set
  (void)hipFuncSetAttribute((const void*)&gru_persistent,
                            hipFuncAttributeMaxDynamicSharedMemorySize, WPK_BYTES);

  // barrier must be 0 at persistent-kernel start on every (replayed) call
  init_barrier<<<1, 1, 0, stream>>>((unsigned*)(ws + OFF_BAR));
  gru_persistent<<<NBLK, NTHREADS, WPK_BYTES, stream>>>(
      x, emb, W_ih, W_hh, b_ih, b_hh, fc_W, fc_b, (float*)d_out, ws);
}
